// Block_79147657331174
// MI455X (gfx1250) — matrix operand order repaired and checked by reading, not compiled, not run on hardware
//
#include <hip/hip_runtime.h>
#include <hip/hip_bf16.h>

// ---------------- types ----------------
typedef _Float16 half_t;
typedef __attribute__((ext_vector_type(16))) _Float16 v16h;
typedef __attribute__((ext_vector_type(8)))  _Float16 v8h;
typedef __attribute__((ext_vector_type(8)))  float    v8f;

#define B_IMG   8
#define NTOK    3136          // 56*56
#define CDIM    512
#define NHEAD   16
#define HDIM    32
#define WS      7
#define WS2     49
#define HID     2048
#define HW      56
#define MROWS   (B_IMG * NTOK)   // 25088 = 128*196

// ---------------- f32 -> f16 convert ----------------
__global__ void f2h_kernel(const float* __restrict__ in, half_t* __restrict__ out, int n) {
    int i = blockIdx.x * blockDim.x + threadIdx.x;
    if (i < n) out[i] = (half_t)in[i];
}

// ---------------- LayerNorm (f32 in -> f16 out) ----------------
__global__ __launch_bounds__(256) void ln_kernel(const float* __restrict__ x,
                                                 const float* __restrict__ g,
                                                 const float* __restrict__ b,
                                                 half_t* __restrict__ out) {
    const int row = blockIdx.x;
    const float* xr = x + (size_t)row * CDIM;
    float s = 0.f, ss = 0.f;
    for (int c = threadIdx.x; c < CDIM; c += 256) {
        float v = xr[c]; s += v; ss += v * v;
    }
    for (int o = 16; o > 0; o >>= 1) { s += __shfl_xor(s, o); ss += __shfl_xor(ss, o); }
    __shared__ float sb[8], ssb[8];
    int wid = threadIdx.x >> 5, lid = threadIdx.x & 31;
    if (lid == 0) { sb[wid] = s; ssb[wid] = ss; }
    __syncthreads();
    if (threadIdx.x == 0) {
        float t = 0.f, tt = 0.f;
        for (int i = 0; i < 8; i++) { t += sb[i]; tt += ssb[i]; }
        sb[0] = t; ssb[0] = tt;
    }
    __syncthreads();
    float mu  = sb[0] * (1.f / CDIM);
    float var = ssb[0] * (1.f / CDIM) - mu * mu;
    float rs  = rsqrtf(var + 1e-5f);
    for (int c = threadIdx.x; c < CDIM; c += 256)
        out[(size_t)row * CDIM + c] = (half_t)((xr[c] - mu) * rs * g[c] + b[c]);
}

// concat two v8h -> v16h
#define CAT16(lo, hi) __builtin_shufflevector(lo, hi, 0,1,2,3,4,5,6,7,8,9,10,11,12,13,14,15)

// ---------------- WMMA GEMM: O[M,N] = A[M,K](f16) * W[N,K]^T(f16) + bias (+res) ----------------
// block = 256 threads = 8 waves; block tile 128x128; wave tile 16x128 (8 x wmma_f32_16x16x32_f16)
template<bool OUT_F16, bool HAS_RES>
__global__ __launch_bounds__(256) void gemm_wmma_kernel(const half_t* __restrict__ A,
                                                        const half_t* __restrict__ W,
                                                        const float*  __restrict__ bias,
                                                        const float*  __restrict__ res,
                                                        void* __restrict__ Out,
                                                        int Nn, int Kk) {
    const int lane = threadIdx.x & 31;
    const int wave = threadIdx.x >> 5;
    const int m0   = blockIdx.y * 128 + wave * 16;
    const int n0   = blockIdx.x * 128;
    const int mr   = lane & 15;
    const int hi   = (lane >> 4) & 1;

    v8f acc[8] = {};
    const half_t* arow = A + (size_t)(m0 + mr) * Kk;

    for (int k0 = 0; k0 < Kk; k0 += 32) {
        // A fragment: lane holds row (m0+mr); halves 0..7 -> K = k0+8*hi.. ; 8..15 -> K = k0+16+8*hi..
        v8h alo = *(const v8h*)(arow + k0 + hi * 8);
        v8h ahi = *(const v8h*)(arow + k0 + 16 + hi * 8);
        v16h afrag = CAT16(alo, ahi);
        if (k0 + 32 < Kk) __builtin_prefetch(arow + k0 + 32, 0, 1);   // global_prefetch_b8

        #pragma unroll
        for (int ni = 0; ni < 8; ni++) {
            // B fragment: lane holds col (n0+ni*16+mr); halves 0..7 -> K = k0+8*hi.. ; 8..15 -> K = k0+16+8*hi..
            const half_t* wp = W + (size_t)(n0 + ni * 16 + mr) * Kk + k0 + hi * 8;
            v16h bfrag = CAT16(*(const v8h*)wp, *(const v8h*)(wp + 16));
            acc[ni] = __builtin_amdgcn_wmma_f32_16x16x32_f16(
                false, afrag, false, bfrag, (short)0, acc[ni], false, false);
        }
    }

    #pragma unroll
    for (int ni = 0; ni < 8; ni++) {
        int col = n0 + ni * 16 + mr;
        float bv = bias[col];
        #pragma unroll
        for (int r = 0; r < 8; r++) {
            int row = m0 + r + hi * 8;
            size_t off = (size_t)row * Nn + col;
            float v = acc[ni][r] + bv;
            if (HAS_RES) v += res[off];
            if (OUT_F16) ((half_t*)Out)[off] = (half_t)v;
            else         ((float*) Out)[off] = v;
        }
    }
}

// ---------------- vectorized LDS fragment loaders (two ds_load_b128 each) ----------------
// A fragment from row-major [16 x K] tile (rowStride halves, multiple of 8)
__device__ inline v16h lds_A_frag(const half_t* p, int rowStride, int lane) {
    int m = lane & 15, hi = (lane >> 4) & 1;
    const half_t* q = p + m * rowStride + hi * 8;
    return CAT16(*(const v8h*)q, *(const v8h*)(q + 16));
}
// B fragment from row-major [N x 32] tile laid out as [n][k] (rowStride halves):
// lane holds col n = lane&15; halves 0..7 -> k = 8*hi + j, 8..15 -> k = 16 + 8*hi + (j-8)
__device__ inline v16h lds_B_frag(const half_t* p, int rowStride, int lane) {
    int n = lane & 15, hi = (lane >> 4) & 1;
    const half_t* q = p + n * rowStride + hi * 8;
    return CAT16(*(const v8h*)q, *(const v8h*)(q + 16));
}

// ---------------- windowed attention: one block per (window, head), 4 waves = 4 row-tiles ----
__global__ __launch_bounds__(128) void attn_kernel(const half_t* __restrict__ qkv,
                                                   const float*  __restrict__ rpb,
                                                   const float*  __restrict__ x,
                                                   float* __restrict__ x1) {
    const int tid  = threadIdx.x;
    const int lane = tid & 31;
    const int mi   = tid >> 5;               // wave index == row-tile index
    const int head = blockIdx.x & (NHEAD - 1);
    const int w    = blockIdx.x >> 4;
    const int b    = w >> 6;                 // 64 windows per image
    const int ww   = w & 63;
    const int wy   = ww >> 3, wx = ww & 7;
    const int m    = lane & 15;
    const int hi   = (lane >> 4) & 1;
    const float scale = 0.17677669529663687f;   // 32^-0.5

    __shared__ __align__(16) half_t Qs[64 * 32];       // [tok][d]
    __shared__ __align__(16) half_t Ks[64 * 32];       // [tok][d]
    __shared__ __align__(16) half_t Vt[32 * 64];       // [d][tok]  (transposed)
    __shared__ __align__(16) half_t Ps[4][16 * 64];    // per-wave P tile
    __shared__ float rpbs[(2 * WS - 1) * (2 * WS - 1)];// 169 floats, this head
    __shared__ float Bs[64 * 64];                      // bias + mask tile, pre-baked

    // stage rel-pos bias for this head
    for (int i = tid; i < 169; i += 128) rpbs[i] = rpb[i * NHEAD + head];

    // stage Q/K (row-major) and V (transposed); 2 threads per token (16 d each)
    {
        int tok = tid >> 1, dh = (tid & 1) * 16;
        bool valid = tok < WS2;
        int iy = tok / WS, ix = tok - (tok / WS) * WS;
        int n = valid ? ((wy * WS + iy) * HW + wx * WS + ix) : 0;
        const half_t* base = qkv + (size_t)(b * NTOK + n) * (3 * CDIM) + head * HDIM + dh;
        v8h z = {};
        #pragma unroll
        for (int d8 = 0; d8 < 16; d8 += 8) {
            v8h qv = valid ? *(const v8h*)(base + d8)            : z;
            v8h kv = valid ? *(const v8h*)(base + CDIM + d8)     : z;
            v8h vv = valid ? *(const v8h*)(base + 2 * CDIM + d8) : z;
            *(v8h*)&Qs[tok * 32 + dh + d8] = qv;
            *(v8h*)&Ks[tok * 32 + dh + d8] = kv;
            #pragma unroll
            for (int t = 0; t < 8; t++) Vt[(dh + d8 + t) * 64 + tok] = vv[t];
        }
    }
    __syncthreads();   // rpbs ready

    // bake 64x64 bias+mask tile: Bs[i][j] = rpb(i,j) if valid else -3e38
    #pragma unroll 4
    for (int e = tid; e < 64 * 64; e += 128) {
        int i = e >> 6, j = e & 63;
        float v = -3.0e38f;
        if (i < WS2 && j < WS2) {
            int iy = i / WS, ix = i - (i / WS) * WS;
            int jy = j / WS, jx = j - (j / WS) * WS;
            v = rpbs[(iy - jy + WS - 1) * (2 * WS - 1) + (ix - jx + WS - 1)];
        }
        Bs[e] = v;
    }
    __syncthreads();   // Qs/Ks/Vt/Bs ready

    // ---- each wave: S tile (16 x 64) = Q[mi] K^T ----
    v8f s[4] = {};
    v16h aq = lds_A_frag(Qs + mi * 16 * 32, 32, lane);
    #pragma unroll
    for (int ni = 0; ni < 4; ni++) {
        v16h bk = lds_B_frag(Ks + ni * 16 * 32, 32, lane);
        s[ni] = __builtin_amdgcn_wmma_f32_16x16x32_f16(
            false, aq, false, bk, (short)0, s[ni], false, false);
    }

    // unconditional logits: mask & bias baked into Bs
    float logit[4][8];
    #pragma unroll
    for (int r = 0; r < 8; r++) {
        int i = mi * 16 + hi * 8 + r;
        #pragma unroll
        for (int ni = 0; ni < 4; ni++)
            logit[ni][r] = fmaf(s[ni][r], scale, Bs[i * 64 + ni * 16 + m]);
    }

    // masked softmax: row i lives in one 16-lane half across the 4 ni regs
    half_t* Pw = Ps[mi];
    #pragma unroll
    for (int r = 0; r < 8; r++) {
        float mx = fmaxf(fmaxf(logit[0][r], logit[1][r]), fmaxf(logit[2][r], logit[3][r]));
        for (int o = 1; o < 16; o <<= 1) mx = fmaxf(mx, __shfl_xor(mx, o));
        float sm = 0.f;
        #pragma unroll
        for (int ni = 0; ni < 4; ni++) { logit[ni][r] = __expf(logit[ni][r] - mx); sm += logit[ni][r]; }
        for (int o = 1; o < 16; o <<= 1) sm += __shfl_xor(sm, o);
        float inv = 1.f / sm;
        #pragma unroll
        for (int ni = 0; ni < 4; ni++)
            Pw[(r + hi * 8) * 64 + ni * 16 + m] = (half_t)(logit[ni][r] * inv);
    }
    __syncthreads();   // uniform across waves; makes P visible for the frag reads

    // ---- O tile (16 x 32) = P (16x64) * V (64x32); V fragments from transposed tile ----
    v8f o[2] = {};
    #pragma unroll
    for (int kt = 0; kt < 2; kt++) {
        v16h ap = lds_A_frag(Pw + kt * 32, 64, lane);
        #pragma unroll
        for (int ni = 0; ni < 2; ni++) {
            // B(k=tok', n=d) = Vt[(ni*16+n)*64 + kt*32 + k], k in table order (8*hi + j, 16 + 8*hi + j-8)
            v16h bv = lds_B_frag(Vt + ni * 16 * 64 + kt * 32, 64, lane);
            o[ni] = __builtin_amdgcn_wmma_f32_16x16x32_f16(
                false, ap, false, bv, (short)0, o[ni], false, false);
        }
    }

    // window-reverse scatter + residual
    #pragma unroll
    for (int r = 0; r < 8; r++) {
        int i = mi * 16 + hi * 8 + r;
        if (i < WS2) {
            int iy = i / WS, ix = i - (i / WS) * WS;
            int n = (wy * WS + iy) * HW + wx * WS + ix;
            #pragma unroll
            for (int ni = 0; ni < 2; ni++) {
                int c = head * HDIM + ni * 16 + m;
                size_t off = (size_t)(b * NTOK + n) * CDIM + c;
                x1[off] = x[off] + o[ni][r];
            }
        }
    }
}

// ---------------- depthwise 3x3 (NHWC) + bias + exact GELU ----------------
__global__ __launch_bounds__(256) void dwconv_gelu_kernel(const half_t* __restrict__ u,
                                                          const float* __restrict__ wts,
                                                          const float* __restrict__ bias,
                                                          half_t* __restrict__ g) {
    long idx = (long)blockIdx.x * 256 + threadIdx.x;
    if (idx >= (long)MROWS * HID) return;
    int c = idx & (HID - 1);
    long rest = idx >> 11;
    int xc = rest % HW; rest /= HW;
    int yc = rest % HW;
    int bi = rest / HW;
    float acc = bias[c];
    #pragma unroll
    for (int dy = -1; dy <= 1; dy++)
        #pragma unroll
        for (int dx = -1; dx <= 1; dx++) {
            int yy = yc + dy, xx = xc + dx;
            if (yy >= 0 && yy < HW && xx >= 0 && xx < HW) {
                size_t off = ((size_t)bi * NTOK + yy * HW + xx) * HID + c;
                acc += (float)u[off] * wts[c * 9 + (dy + 1) * 3 + (dx + 1)];
            }
        }
    g[idx] = (half_t)(0.5f * acc * (1.f + erff(acc * 0.70710678118654752f)));
}

// ---------------- host launcher ----------------
static inline size_t alignup(size_t v) { return (v + 255) & ~(size_t)255; }

extern "C" void kernel_launch(void* const* d_in, const int* in_sizes, int n_in,
                              void* d_out, int out_size, void* d_ws, size_t ws_size,
                              hipStream_t stream) {
    const float* x      = (const float*)d_in[0];
    const float* n1g    = (const float*)d_in[3];
    const float* n1b    = (const float*)d_in[4];
    const float* n2g    = (const float*)d_in[5];
    const float* n2b    = (const float*)d_in[6];
    const float* qkv_w  = (const float*)d_in[7];
    const float* qkv_b  = (const float*)d_in[8];
    const float* rpb    = (const float*)d_in[9];
    const float* fc1_w  = (const float*)d_in[10];
    const float* fc1_b  = (const float*)d_in[11];
    const float* dw_w   = (const float*)d_in[12];
    const float* dw_b   = (const float*)d_in[13];
    const float* fc2_w  = (const float*)d_in[14];
    const float* fc2_b  = (const float*)d_in[15];

    char* p = (char*)d_ws;
    half_t* h16   = (half_t*)p; p += alignup((size_t)MROWS * CDIM * 2);
    half_t* qkv16 = (half_t*)p; p += alignup((size_t)MROWS * 3 * CDIM * 2);
    float*  x1    = (float*) p; p += alignup((size_t)MROWS * CDIM * 4);
    half_t* u16   = (half_t*)p; p += alignup((size_t)MROWS * HID * 2);
    half_t* g16   = (half_t*)p; p += alignup((size_t)MROWS * HID * 2);
    half_t* wq16  = (half_t*)p; p += alignup((size_t)3 * CDIM * CDIM * 2);
    half_t* w116  = (half_t*)p; p += alignup((size_t)HID * CDIM * 2);
    half_t* w216  = (half_t*)p; p += alignup((size_t)CDIM * HID * 2);

    // weights -> f16
    f2h_kernel<<<(3 * CDIM * CDIM + 255) / 256, 256, 0, stream>>>(qkv_w, wq16, 3 * CDIM * CDIM);
    f2h_kernel<<<(HID * CDIM + 255) / 256, 256, 0, stream>>>(fc1_w, w116, HID * CDIM);
    f2h_kernel<<<(CDIM * HID + 255) / 256, 256, 0, stream>>>(fc2_w, w216, CDIM * HID);

    // LN1 + QKV projection
    ln_kernel<<<MROWS, 256, 0, stream>>>(x, n1g, n1b, h16);
    gemm_wmma_kernel<true, false><<<dim3(3 * CDIM / 128, MROWS / 128), 256, 0, stream>>>(
        h16, wq16, qkv_b, nullptr, qkv16, 3 * CDIM, CDIM);

    // windowed attention + residual (1 block per window-head, 4 waves)
    attn_kernel<<<B_IMG * 64 * NHEAD, 128, 0, stream>>>(qkv16, rpb, x, x1);

    // LN2 + FC1
    ln_kernel<<<MROWS, 256, 0, stream>>>(x1, n2g, n2b, h16);
    gemm_wmma_kernel<true, false><<<dim3(HID / 128, MROWS / 128), 256, 0, stream>>>(
        h16, w116, fc1_b, nullptr, u16, HID, CDIM);

    // depthwise conv + GELU
    dwconv_gelu_kernel<<<(unsigned)(((long)MROWS * HID + 255) / 256), 256, 0, stream>>>(
        u16, dw_w, dw_b, g16);

    // FC2 + residual -> d_out (f32)
    gemm_wmma_kernel<false, true><<<dim3(CDIM / 128, MROWS / 128), 256, 0, stream>>>(
        g16, w216, fc2_b, x1, (float*)d_out, CDIM, HID);
}
